// GINO_Decoder_33071248179439
// MI455X (gfx1250) — matrix operand order repaired and checked by reading, not compiled, not run on hardware
//
#include <hip/hip_runtime.h>
#include <hip/hip_bf16.h>

// ---------------------------------------------------------------------------
// GINO decoder for MI455X (gfx1250, wave32, WMMA).
// Problem constants (from the reference): n_in = 32^3 = 32768 latent points,
// n_out = 16384 queries, C = 256, hidden = 512, D = 3, BATCH = 2, OUT_CH = 4.
// ---------------------------------------------------------------------------

typedef _Float16 v16h __attribute__((ext_vector_type(16)));
typedef _Float16 v8h  __attribute__((ext_vector_type(8)));
typedef float    v8f  __attribute__((ext_vector_type(8)));

#define N_IN   32768
#define N_OUT  16384
#define CCH    256
#define H1DIM  512
#define OUTCH  4

__device__ __forceinline__ float gelu_exact(float x) {
    return 0.5f * x * (1.0f + erff(x * 0.70710678118654752440f));
}

// ---------------------------------------------------------------------------
// Kernel 0: convert W2 (512x256) and W3 (256x256) to f16, transposed to
// N-major (W2T: 256x512, W3T: 256x256) so each lane's WMMA B-fragment is a
// contiguous 32B run -> global_load_b128 pairs.
// ---------------------------------------------------------------------------
__global__ __launch_bounds__(256) void prep_weights(
    const float* __restrict__ W2, const float* __restrict__ W3,
    _Float16* __restrict__ W2T, _Float16* __restrict__ W3T) {
    int idx = blockIdx.x * blockDim.x + threadIdx.x;
    if (idx < H1DIM * CCH) {                       // W2 is (512, 256) row-major
        int k = idx / CCH, n = idx % CCH;
        W2T[n * H1DIM + k] = (_Float16)W2[idx];
    } else {
        int j = idx - H1DIM * CCH;
        if (j < CCH * CCH) {                       // W3 is (256, 256) row-major
            int k = j / CCH, n = j % CCH;
            W3T[n * CCH + k] = (_Float16)W3[j];
        }
    }
}

// ---------------------------------------------------------------------------
// Kernel 1: per-edge MLP, 16 edges per wave32, fused message + atomic scatter.
//
// WMMA fragment layouts used here (contiguous k order, same for A and B):
//   A (16x32 f16): lane&15 = M;  lane<16 -> K = k0+0..15
//                                lane>=16 -> K = k0+16..31
//   B (32x16 f16): lane&15 = N;  lane<16 -> K = k0+0..15, lane>=16 -> K = k0+16..31
//   C/D (16x16 f32): vgpr v, lane -> element [M = v + 8*(lane>>4)][N = lane&15]
// ---------------------------------------------------------------------------
__global__ __launch_bounds__(32) void edge_mlp(
    const float* __restrict__ f_y,     // (2, 32768, 256)
    const float* __restrict__ yq,      // latent grid coords (32768, 3)
    const float* __restrict__ xq,      // output queries     (16384, 3)
    const int*   __restrict__ nb_index,
    const int*   __restrict__ out_index,
    const float* __restrict__ W1,      // (6, 512)  f32
    const float* __restrict__ b1,      // (512)
    const float* __restrict__ b2,      // (256)
    const float* __restrict__ b3,      // (256)
    const _Float16* __restrict__ W2T,  // (256, 512) f16 (N-major)
    const _Float16* __restrict__ W3T,  // (256, 256) f16 (N-major)
    float* __restrict__ sums,          // (2, 16384, 256)  accum
    float* __restrict__ cnt,           // (16384)
    int E) {

    __shared__ __align__(32) _Float16 sh1[16][H1DIM];  // 16 KB
    __shared__ __align__(32) _Float16 sh2[16][CCH];    //  8 KB
    __shared__ float s_agg[16][6];
    __shared__ int   s_nb[16];
    __shared__ int   s_oi[16];

    const int lane     = threadIdx.x;        // 0..31
    const int m        = lane & 15;          // edge row within tile / A row
    const int half_sel = lane >> 4;          // 0 or 1 (lane half)
    const int nl       = lane & 15;          // B/D column within tile
    const int tile0    = blockIdx.x * 16;    // first edge of this tile

    // ---- stage 16 edges of metadata + 6 input features -------------------
    if (lane < 16) {
        int e      = tile0 + lane;
        bool valid = (e < E);
        int ec     = valid ? e : (E - 1);
        int nb     = nb_index[ec];
        int oi     = out_index[ec];
        s_nb[lane] = nb;
        s_oi[lane] = valid ? oi : -1;
        s_agg[lane][0] = yq[nb * 3 + 0];
        s_agg[lane][1] = yq[nb * 3 + 1];
        s_agg[lane][2] = yq[nb * 3 + 2];
        s_agg[lane][3] = xq[oi * 3 + 0];
        s_agg[lane][4] = xq[oi * 3 + 1];
        s_agg[lane][5] = xq[oi * 3 + 2];
        if (valid) atomicAdd(&cnt[oi], 1.0f);
    }
    __syncthreads();

    // ---- layer 1: (16 x 6) @ (6 x 512), K=6 -> plain VALU FMA ------------
    // lanes 0..15 cover m x n[0..255], lanes 16..31 cover m x n[256..511]
    {
        float a0 = s_agg[m][0], a1 = s_agg[m][1], a2 = s_agg[m][2];
        float a3 = s_agg[m][3], a4 = s_agg[m][4], a5 = s_agg[m][5];
        int nbase = half_sel * 256;
        for (int n = nbase; n < nbase + 256; ++n) {
            float acc = b1[n];
            acc = fmaf(a0, W1[0 * H1DIM + n], acc);
            acc = fmaf(a1, W1[1 * H1DIM + n], acc);
            acc = fmaf(a2, W1[2 * H1DIM + n], acc);
            acc = fmaf(a3, W1[3 * H1DIM + n], acc);
            acc = fmaf(a4, W1[4 * H1DIM + n], acc);
            acc = fmaf(a5, W1[5 * H1DIM + n], acc);
            sh1[m][n] = (_Float16)gelu_exact(acc);
        }
    }
    __syncthreads();

    const int kbA = half_sel * 16;   // A-fragment K base within 32-chunk
    const int kbB = half_sel * 16;   // B-fragment K base within 32-chunk

    // ---- layer 2: (16 x 512) @ (512 x 256) via WMMA f16->f32 -------------
    for (int nt = 0; nt < CCH / 16; ++nt) {
        const int ng = nt * 16 + nl;
        const _Float16* __restrict__ brow = W2T + ng * H1DIM;
        v8f c = {};
        for (int ks = 0; ks < H1DIM / 32; ++ks) {
            const int k0 = ks * 32;
            v8h lo = *(const v8h*)&sh1[m][k0 + kbA];
            v8h hi = *(const v8h*)&sh1[m][k0 + kbA + 8];
            v16h a;
#pragma unroll
            for (int j = 0; j < 8; ++j) { a[j] = lo[j]; a[j + 8] = hi[j]; }
            v16h b = *(const v16h*)(brow + k0 + kbB);
            c = __builtin_amdgcn_wmma_f32_16x16x32_f16(
                    false, a, false, b, (short)0, c, false, false);
        }
        const float bias = b2[ng];
#pragma unroll
        for (int v = 0; v < 8; ++v) {
            int mm = v + 8 * half_sel;
            sh2[mm][ng] = (_Float16)gelu_exact(c[v] + bias);
        }
    }
    __syncthreads();

    // ---- layer 3: (16 x 256) @ (256 x 256) via WMMA, fused message -------
    for (int nt = 0; nt < CCH / 16; ++nt) {
        const int ng = nt * 16 + nl;
        const _Float16* __restrict__ brow = W3T + ng * CCH;
        v8f c = {};
        for (int ks = 0; ks < CCH / 32; ++ks) {
            const int k0 = ks * 32;
            v8h lo = *(const v8h*)&sh2[m][k0 + kbA];
            v8h hi = *(const v8h*)&sh2[m][k0 + kbA + 8];
            v16h a;
#pragma unroll
            for (int j = 0; j < 8; ++j) { a[j] = lo[j]; a[j + 8] = hi[j]; }
            v16h b = *(const v16h*)(brow + k0 + kbB);
            c = __builtin_amdgcn_wmma_f32_16x16x32_f16(
                    false, a, false, b, (short)0, c, false, false);
        }
        const float bias = b3[ng];
#pragma unroll
        for (int v = 0; v < 8; ++v) {
            int mm = v + 8 * half_sel;           // edge row within tile
            int oi = s_oi[mm];
            if (oi >= 0) {                       // divergence only after WMMA
                float kval = c[v] + bias;        // kernel value k(x_i, y_j)[ng]
                int   nb   = s_nb[mm];
                float f0 = f_y[(0 * N_IN + nb) * CCH + ng];
                float f1 = f_y[(1 * N_IN + nb) * CCH + ng];
                atomicAdd(&sums[(0 * N_OUT + oi) * CCH + ng], kval * f0);
                atomicAdd(&sums[(1 * N_OUT + oi) * CCH + ng], kval * f1);
            }
        }
    }
}

// ---------------------------------------------------------------------------
// Kernel 2: mean (divide by count) + 256->4 projection per query.
// One 256-thread block per query; thread = channel; LDS tree reduce.
// ---------------------------------------------------------------------------
__global__ __launch_bounds__(256) void finalize(
    const float* __restrict__ sums, const float* __restrict__ cnt,
    const float* __restrict__ Wp,   const float* __restrict__ bp,
    float* __restrict__ out) {
    const int i = blockIdx.x;
    const int c = threadIdx.x;

    __shared__ float part[256][8];

    const float inv = 1.0f / fmaxf(cnt[i], 1.0f);
    const float m0  = sums[(0 * N_OUT + i) * CCH + c] * inv;
    const float m1  = sums[(1 * N_OUT + i) * CCH + c] * inv;
#pragma unroll
    for (int oc = 0; oc < OUTCH; ++oc) {
        float w = Wp[c * OUTCH + oc];
        part[c][oc]         = m0 * w;
        part[c][OUTCH + oc] = m1 * w;
    }
    __syncthreads();
    for (int s = 128; s > 0; s >>= 1) {
        if (c < s) {
#pragma unroll
            for (int j = 0; j < 8; ++j) part[c][j] += part[c + s][j];
        }
        __syncthreads();
    }
    if (c < 8) {
        int b  = c >> 2;
        int oc = c & 3;
        out[(b * N_OUT + i) * OUTCH + oc] = part[0][c] + bp[oc];
    }
}

// ---------------------------------------------------------------------------
// Host-side launcher.  Inputs (setup_inputs order):
//  0 latent_embed (2,32,32,32,256) f32   1 latent_queries (1,32,32,32,3) f32
//  2 output_queries (1,16384,3) f32      3 nb_index (E) i32   4 out_index (E) i32
//  5 W1 (6,512)  6 b1 (512)  7 W2 (512,256)  8 b2 (256)
//  9 W3 (256,256) 10 b3 (256) 11 Wp (256,4)  12 bp (4)        all f32
// ---------------------------------------------------------------------------
extern "C" void kernel_launch(void* const* d_in, const int* in_sizes, int n_in_args,
                              void* d_out, int out_size, void* d_ws, size_t ws_size,
                              hipStream_t stream) {
    const float* f_y  = (const float*)d_in[0];
    const float* latq = (const float*)d_in[1];
    const float* outq = (const float*)d_in[2];
    const int*   nbi  = (const int*)d_in[3];
    const int*   oii  = (const int*)d_in[4];
    const float* W1   = (const float*)d_in[5];
    const float* b1   = (const float*)d_in[6];
    const float* W2   = (const float*)d_in[7];
    const float* b2   = (const float*)d_in[8];
    const float* W3   = (const float*)d_in[9];
    const float* b3   = (const float*)d_in[10];
    const float* Wp   = (const float*)d_in[11];
    const float* bp   = (const float*)d_in[12];
    float* out        = (float*)d_out;

    const int E = in_sizes[3];

    // workspace layout (bytes):
    //   [0, 32MB)        sums  (2,16384,256) f32
    //   [32MB, +64KB)    cnt   (16384) f32
    //   then W2T (256x512 f16), W3T (256x256 f16)
    char* ws = (char*)d_ws;
    const size_t OFF_SUMS = 0;
    const size_t OFF_CNT  = (size_t)2 * N_OUT * CCH * sizeof(float);      // 33554432
    const size_t OFF_W2T  = OFF_CNT + (size_t)N_OUT * sizeof(float);      // 33619968
    const size_t OFF_W3T  = OFF_W2T + (size_t)CCH * H1DIM * sizeof(_Float16);
    float*     sums = (float*)(ws + OFF_SUMS);
    float*     cnt  = (float*)(ws + OFF_CNT);
    _Float16*  W2T  = (_Float16*)(ws + OFF_W2T);
    _Float16*  W3T  = (_Float16*)(ws + OFF_W3T);

    // zero the accumulators every call (graph-replay safe; memset node)
    hipMemsetAsync(sums, 0, OFF_CNT + (size_t)N_OUT * sizeof(float) - OFF_SUMS, stream);

    // weight prep: 512*256 + 256*256 = 196608 elements
    prep_weights<<<(H1DIM * CCH + CCH * CCH + 255) / 256, 256, 0, stream>>>(
        W2, W3, W2T, W3T);

    if (E > 0) {
        int ntiles = (E + 15) / 16;
        edge_mlp<<<ntiles, 32, 0, stream>>>(
            f_y, latq, outq, nbi, oii, W1, b1, b2, b3, W2T, W3T, sums, cnt, E);
    }

    finalize<<<N_OUT, 256, 0, stream>>>(sums, cnt, Wp, bp, out);
}
